// TransformerDecoderLayer_40948218200418
// MI455X (gfx1250) — compile-verified, matrix operand order checked by reading, not run on hardware
//
#include <hip/hip_runtime.h>
#include <hip/hip_bf16.h>
#include <math.h>

// ---------------------------------------------------------------------------
// Types for CDNA5 WMMA (wave32): v_wmma_f32_16x16x32_bf16
// ---------------------------------------------------------------------------
typedef __bf16 bf16;
typedef __bf16 bfv8  __attribute__((ext_vector_type(8)));
typedef __bf16 bfv16 __attribute__((ext_vector_type(16)));
typedef float  fv4   __attribute__((ext_vector_type(4)));
typedef float  fv8   __attribute__((ext_vector_type(8)));

#define GBM 128
#define GBN 128
#define GBK 32
#define LDSK 40   // padded LDS row stride (bf16 elems) to dodge bank conflicts

// Generic batched GEMM:  C[bi] = act( (A[bi] @ W[bi]^T + bias) * alpha )
// A is MxK (row stride lda), W is NxK (row stride ldw) -> both K-contiguous.
// flags: 1=relu, 2=sigmoid, 4=accumulate into fp32 C, 8=store bf16
// Block: 256 threads = 8 wave32 waves; each wave owns a 32x64 patch
// (2x4 WMMA tiles, 8 f32 accumulators) of the 128x128 block tile.
template<int ABF, int WBF>
__global__ __launch_bounds__(256) void gemm_k(
    const void* __restrict__ Av, const void* __restrict__ Wv,
    const float* __restrict__ bias, void* __restrict__ Cv,
    int M, int N, int K,
    long lda, long ldw, long ldc, long bsA, long bsW, long bsC,
    int flags, float alpha)
{
  __shared__ bf16 As[GBM * LDSK];
  __shared__ bf16 Ws[GBN * LDSK];
  const int tid  = threadIdx.x;
  const int lane = tid & 31;
  const int wid  = tid >> 5;
  const int wm = wid & 3, wn = wid >> 2;          // 4 waves along M, 2 along N
  const int m0 = blockIdx.y * GBM;
  const int n0 = blockIdx.x * GBN;
  const long bi = blockIdx.z;

  fv8 acc[2][4];
  #pragma unroll
  for (int i = 0; i < 2; i++)
    #pragma unroll
    for (int j = 0; j < 4; j++)
      #pragma unroll
      for (int e = 0; e < 8; e++) acc[i][j][e] = 0.0f;

  // staging: 256 threads, each thread moves 16 contiguous elements of one row
  const int srow = tid >> 1, scol = (tid & 1) * 16;   // 128 rows x 32 cols
  const int fr = lane & 15;          // row/col within 16x16 tile
  const int kb = (lane >> 4) * 8;    // K base for lane half (ISA layout)

  for (int kt = 0; kt < K; kt += GBK) {
    { // stage A tile -> bf16 LDS
      bf16* dst = &As[srow * LDSK + scol];
      int gr = m0 + srow;
      if (gr < M) {
        if (ABF) {
          const bf16* src = (const bf16*)Av + bi * bsA + (long)gr * lda + kt + scol;
          *(bfv8*)dst       = *(const bfv8*)src;
          *(bfv8*)(dst + 8) = *(const bfv8*)(src + 8);
          if (kt + GBK < K) __builtin_prefetch(src + GBK, 0, 1);  // global_prefetch_b8
        } else {
          const float* src = (const float*)Av + bi * bsA + (long)gr * lda + kt + scol;
          #pragma unroll
          for (int v = 0; v < 4; v++) {
            fv4 x = *(const fv4*)(src + v * 4);
            dst[v*4+0] = (bf16)x[0]; dst[v*4+1] = (bf16)x[1];
            dst[v*4+2] = (bf16)x[2]; dst[v*4+3] = (bf16)x[3];
          }
          if (kt + GBK < K) __builtin_prefetch(src + GBK, 0, 1);
        }
      } else {
        #pragma unroll
        for (int i2 = 0; i2 < 16; i2++) dst[i2] = (bf16)0.0f;
      }
    }
    { // stage W tile -> bf16 LDS
      bf16* dst = &Ws[srow * LDSK + scol];
      int gn = n0 + srow;
      if (gn < N) {
        if (WBF) {
          const bf16* src = (const bf16*)Wv + bi * bsW + (long)gn * ldw + kt + scol;
          *(bfv8*)dst       = *(const bfv8*)src;
          *(bfv8*)(dst + 8) = *(const bfv8*)(src + 8);
          if (kt + GBK < K) __builtin_prefetch(src + GBK, 0, 1);
        } else {
          const float* src = (const float*)Wv + bi * bsW + (long)gn * ldw + kt + scol;
          #pragma unroll
          for (int v = 0; v < 4; v++) {
            fv4 x = *(const fv4*)(src + v * 4);
            dst[v*4+0] = (bf16)x[0]; dst[v*4+1] = (bf16)x[1];
            dst[v*4+2] = (bf16)x[2]; dst[v*4+3] = (bf16)x[3];
          }
          if (kt + GBK < K) __builtin_prefetch(src + GBK, 0, 1);
        }
      } else {
        #pragma unroll
        for (int i2 = 0; i2 < 16; i2++) dst[i2] = (bf16)0.0f;
      }
    }
    __syncthreads();

    // Fragment loads per ISA 16-bit A layout: lane holds row (lane&15),
    // elements 0..7 -> K = kb..kb+7, elements 8..15 -> K = kb+16..kb+23.
    bfv16 af[2], bfr[4];
    #pragma unroll
    for (int i = 0; i < 2; i++) {
      const bf16* p = &As[(wm * 32 + i * 16 + fr) * LDSK + kb];
      bfv8 lo = *(const bfv8*)p;
      bfv8 hi = *(const bfv8*)(p + 16);
      #pragma unroll
      for (int e = 0; e < 8; e++) { af[i][e] = lo[e]; af[i][8 + e] = hi[e]; }
    }
    #pragma unroll
    for (int j = 0; j < 4; j++) {
      const bf16* p = &Ws[(wn * 64 + j * 16 + fr) * LDSK + kb];
      bfv8 lo = *(const bfv8*)p;
      bfv8 hi = *(const bfv8*)(p + 16);
      #pragma unroll
      for (int e = 0; e < 8; e++) { bfr[j][e] = lo[e]; bfr[j][8 + e] = hi[e]; }
    }
    #pragma unroll
    for (int i = 0; i < 2; i++)
      #pragma unroll
      for (int j = 0; j < 4; j++)
        acc[i][j] = __builtin_amdgcn_wmma_f32_16x16x32_bf16(
            false, af[i], false, bfr[j], (short)0, acc[i][j], false, false);
    __syncthreads();
  }

  // C layout: lane holds col (lane&15); rows = (lane>>4)*8 + r over 8 VGPRs
  #pragma unroll
  for (int j = 0; j < 4; j++) {
    int col = n0 + wn * 64 + j * 16 + fr;
    if (col >= N) continue;
    float bv = bias ? bias[col] : 0.0f;
    #pragma unroll
    for (int i = 0; i < 2; i++) {
      int rtop = m0 + wm * 32 + i * 16 + kb;
      #pragma unroll
      for (int r = 0; r < 8; r++) {
        int row = rtop + r;
        if (row >= M) continue;
        float v = (acc[i][j][r] + bv) * alpha;
        if (flags & 1) v = fmaxf(v, 0.0f);
        if (flags & 2) v = 1.0f / (1.0f + expf(-v));
        long cidx = bi * bsC + (long)row * ldc + col;
        if (flags & 8) {
          ((bf16*)Cv)[cidx] = (bf16)v;
        } else {
          float* Cf = (float*)Cv;
          if (flags & 4) v += Cf[cidx];
          Cf[cidx] = v;
        }
      }
    }
  }
}

// ---------------------------------------------------------------------------
// Elementwise / reduction kernels
// ---------------------------------------------------------------------------
__global__ void vr_k(const unsigned char* __restrict__ kpm,
                     const int* __restrict__ mhp, const int* __restrict__ mwp,
                     float* __restrict__ vr)
{
  int b = threadIdx.x;
  if (b < 8) {
    int mh = *mhp, mw = *mwp;
    int ch = 0, cw = 0;
    for (int y = 0; y < mh; y++) ch += kpm[b * mh * mw + y * mw] ? 0 : 1;
    for (int x = 0; x < mw; x++) cw += kpm[b * mh * mw + x] ? 0 : 1;
    float vH = (float)ch / (float)mh, vW = (float)cw / (float)mw;
    vr[b*4+0] = vW; vr[b*4+1] = vH; vr[b*4+2] = vW; vr[b*4+3] = vH;
  }
}

__global__ __launch_bounds__(256) void ln_k(
    const float* __restrict__ X, const float* __restrict__ Dd,
    const float* __restrict__ g, const float* __restrict__ b,
    float* __restrict__ O)
{
  const long row = blockIdx.x;
  const int tid = threadIdx.x;
  __shared__ float r1[8], r2[8];
  float v = X[row * 256 + tid] + Dd[row * 256 + tid];
  float s = v;
  for (int o = 16; o; o >>= 1) s += __shfl_xor(s, o, 32);
  if ((tid & 31) == 0) r1[tid >> 5] = s;
  __syncthreads();
  s = 0.0f;
  #pragma unroll
  for (int i = 0; i < 8; i++) s += r1[i];
  float mean = s * (1.0f / 256.0f);
  float d = v - mean;
  float q = d * d;
  for (int o = 16; o; o >>= 1) q += __shfl_xor(q, o, 32);
  if ((tid & 31) == 0) r2[tid >> 5] = q;
  __syncthreads();
  q = 0.0f;
  #pragma unroll
  for (int i = 0; i < 8; i++) q += r2[i];
  float var = q * (1.0f / 256.0f);
  O[row * 256 + tid] = d * rsqrtf(var + 1e-5f) * g[tid] + b[tid];
}

// self-attn softmax over 300 keys, write bf16 padded to 320 (zeros in pad)
__global__ __launch_bounds__(256) void softmax_sa_k(
    const float* __restrict__ S, bf16* __restrict__ A)
{
  const long row = blockIdx.x;
  const float* s = S + row * 320;
  bf16* a = A + row * 320;
  const int tid = threadIdx.x;
  __shared__ float r1[8], r2[8];
  float x0 = (tid < 300) ? s[tid] : -3.0e38f;
  float x1 = (tid + 256 < 300) ? s[tid + 256] : -3.0e38f;
  float m = fmaxf(x0, x1);
  for (int o = 16; o; o >>= 1) m = fmaxf(m, __shfl_xor(m, o, 32));
  if ((tid & 31) == 0) r1[tid >> 5] = m;
  __syncthreads();
  m = r1[0];
  #pragma unroll
  for (int i = 1; i < 8; i++) m = fmaxf(m, r1[i]);
  float e0 = (tid < 300) ? expf(x0 - m) : 0.0f;
  float e1 = (tid + 256 < 300) ? expf(x1 - m) : 0.0f;
  float sum = e0 + e1;
  for (int o = 16; o; o >>= 1) sum += __shfl_xor(sum, o, 32);
  if ((tid & 31) == 0) r2[tid >> 5] = sum;
  __syncthreads();
  sum = 0.0f;
  #pragma unroll
  for (int i = 0; i < 8; i++) sum += r2[i];
  float inv = 1.0f / sum;
  a[tid] = (bf16)(e0 * inv);
  if (tid + 256 < 320) a[tid + 256] = (bf16)(e1 * inv);
}

// cross-attn softmax over 1024 keys, fused gaussian bias + key padding mask
__global__ __launch_bounds__(256) void softmax_ca_k(
    const float* __restrict__ S, const float* __restrict__ gp,
    const float* __restrict__ gs2, const float* __restrict__ grid,
    const unsigned char* __restrict__ kpm, bf16* __restrict__ A)
{
  const long rid = blockIdx.x;            // bi*300 + q , bi = b*8+h
  const int bi = (int)(rid / 300), q = (int)(rid % 300);
  const int b = bi >> 3;
  const long prow = (long)q * 64 + bi;
  const float gpx = gp[prow*2], gpy = gp[prow*2+1];
  const float sx = gs2[prow*2], sy = gs2[prow*2+1];
  const float* s = S + rid * 1024;
  bf16* a = A + rid * 1024;
  const int tid = threadIdx.x;
  __shared__ float r1[8], r2[8];
  float x[4];
  float m = -3.0e38f;
  #pragma unroll
  for (int t = 0; t < 4; t++) {
    int k = tid + t * 256;
    float gx = grid[k*2], gy = grid[k*2+1];
    float dx = gpx - (gx + 0.5f), dy = gpy - (gy + 0.5f);
    float gau = -fabsf(dx*dx*sx + dy*dy*sy) * 0.125f;
    float v = s[k] + gau;
    if (kpm[b*1024 + k]) v = -1.0e9f;
    x[t] = v; m = fmaxf(m, v);
  }
  for (int o = 16; o; o >>= 1) m = fmaxf(m, __shfl_xor(m, o, 32));
  if ((tid & 31) == 0) r1[tid >> 5] = m;
  __syncthreads();
  m = r1[0];
  #pragma unroll
  for (int i = 1; i < 8; i++) m = fmaxf(m, r1[i]);
  float sum = 0.0f;
  #pragma unroll
  for (int t = 0; t < 4; t++) { x[t] = expf(x[t] - m); sum += x[t]; }
  for (int o = 16; o; o >>= 1) sum += __shfl_xor(sum, o, 32);
  if ((tid & 31) == 0) r2[tid >> 5] = sum;
  __syncthreads();
  sum = 0.0f;
  #pragma unroll
  for (int i = 0; i < 8; i++) sum += r2[i];
  float inv = 1.0f / sum;
  #pragma unroll
  for (int t = 0; t < 4; t++) a[tid + t * 256] = (bf16)(x[t] * inv);
}

// transpose V head-slices to [bi][d=32][Kpad] bf16 (zero pad past Kvalid)
__global__ __launch_bounds__(256) void vtr_k(
    const float* __restrict__ V, bf16* __restrict__ VT, int Kvalid, int Kpad)
{
  int bi = blockIdx.x; int b = bi >> 3, h = bi & 7;
  int d = threadIdx.x & 31, kk = threadIdx.x >> 5;
  for (int k0 = kk; k0 < Kpad; k0 += 8) {
    float v = (k0 < Kvalid) ? V[((long)k0 * 8 + b) * 256 + h * 32 + d] : 0.0f;
    VT[((long)bi * 32 + d) * Kpad + k0] = (bf16)v;
  }
}

// ROI align: row = ((b*300+q)*49 + py*7+px), 256 threads over channels
__global__ __launch_bounds__(256) void roi_k(
    const float* __restrict__ mem, const float* __restrict__ rb,
    const float* __restrict__ vr, const int* __restrict__ mhp,
    const int* __restrict__ mwp, bf16* __restrict__ roi)
{
  const long rid = blockIdx.x;
  int p = (int)(rid % 49); long nn = rid / 49;
  int q = (int)(nn % 300); int b = (int)(nn / 300);
  int px = p % 7, py = p / 7;
  int mh = *mhp, mw = *mwp;
  const float* box = rb + ((long)b * 300 + q) * 4;
  float cx = box[0], cy = box[1], w = box[2], hh = box[3];
  float x1 = (cx - 0.5f*w)  * mw * vr[b*4+0] - 0.5f;
  float y1 = (cy - 0.5f*hh) * mh * vr[b*4+1] - 0.5f;
  float x2 = (cx + 0.5f*w)  * mw * vr[b*4+2] - 0.5f;
  float y2 = (cy + 0.5f*hh) * mh * vr[b*4+3] - 0.5f;
  float bw = (x2 - x1) / 7.0f, bh = (y2 - y1) / 7.0f;
  float xs = x1 + (px + 0.5f) * bw;
  float ys = y1 + (py + 0.5f) * bh;
  bool valid = (ys > -1.0f) && (ys < (float)mh) && (xs > -1.0f) && (xs < (float)mw);
  float y = fminf(fmaxf(ys, 0.0f), (float)(mh - 1));
  float x = fminf(fmaxf(xs, 0.0f), (float)(mw - 1));
  int y0 = (int)floorf(y), x0 = (int)floorf(x);
  int y1i = min(y0 + 1, mh - 1), x1i = min(x0 + 1, mw - 1);
  float ly = y - y0, lx = x - x0;
  float w00 = (1-ly)*(1-lx), w01 = (1-ly)*lx, w10 = ly*(1-lx), w11 = ly*lx;
  int c = threadIdx.x;
  long base = (long)b * 256 + c;   // memory[(y*mw+x)*8*256 + b*256 + c]
  float v = w00 * mem[(long)(y0 * mw + x0 ) * 2048 + base]
          + w01 * mem[(long)(y0 * mw + x1i) * 2048 + base]
          + w10 * mem[(long)(y1i * mw + x0 ) * 2048 + base]
          + w11 * mem[(long)(y1i * mw + x1i) * 2048 + base];
  roi[rid * 256 + c] = (bf16)(valid ? v : 0.0f);
}

// tanh points (index n*8+h, n = b*300+q), scales^2, q_pos, gp (flat index (q*64 + b*8 + h))
__global__ void points_k(
    const float* __restrict__ pts5, const float* __restrict__ rb,
    const float* __restrict__ vr, const int* __restrict__ mhp,
    const int* __restrict__ mwp,
    float* __restrict__ points, float* __restrict__ qpos,
    float* __restrict__ gp, float* __restrict__ gs2)
{
  long i = (long)blockIdx.x * blockDim.x + threadIdx.x;  // n*8+h, n = b*300+q
  if (i >= 2400L * 8) return;
  int h = (int)(i & 7); long n = i >> 3;
  int q = (int)(n % 300), b = (int)(n / 300);
  float tpx = tanhf(pts5[n*32 + 2*h]);
  float tpy = tanhf(pts5[n*32 + 2*h + 1]);
  points[i*2] = tpx; points[i*2+1] = tpy;
  float sx = pts5[n*32 + 16 + 2*h], sy = pts5[n*32 + 16 + 2*h + 1];
  long pr = (long)q * 64 + b * 8 + h;
  gs2[pr*2] = sx*sx; gs2[pr*2+1] = sy*sy;
  const float* box = rb + n * 4;
  float qx = box[0] + 0.5f * box[2] * tpx;
  float qy = box[1] + 0.5f * box[3] * tpy;
  qpos[pr*2] = qx; qpos[pr*2+1] = qy;
  gp[pr*2]   = qx * (float)(*mwp) * vr[b*4+0];
  gp[pr*2+1] = qy * (float)(*mhp) * vr[b*4+1];
}

// grid_sample on 7x7 roi + sigmoid-gate multiply; row = (q*8+b)*8+h
__global__ __launch_bounds__(256) void sample_k(
    const bf16* __restrict__ roi, const float* __restrict__ pts,
    const float* __restrict__ g1, float* __restrict__ qc)
{
  const long row = blockIdx.x;
  int h = (int)(row & 7); long qb = row >> 3;
  int b = (int)(qb & 7); int q = (int)(qb >> 3);
  long n = (long)b * 300 + q;
  float gx = pts[(n*8 + h)*2], gy = pts[(n*8 + h)*2 + 1];
  float x = ((gx + 1.0f) * 7.0f - 1.0f) * 0.5f;
  float y = ((gy + 1.0f) * 7.0f - 1.0f) * 0.5f;
  int x0 = (int)floorf(x), y0 = (int)floorf(y);
  float lx = x - x0, ly = y - y0;
  int c = threadIdx.x;
  auto tap = [&](int ix, int iy, float w) -> float {
    bool ok = (ix >= 0) && (ix < 7) && (iy >= 0) && (iy < 7);
    int cxi = min(max(ix, 0), 6), cyi = min(max(iy, 0), 6);
    float v = (float)roi[((long)n * 49 + cyi * 7 + cxi) * 256 + c];
    return ok ? w * v : 0.0f;
  };
  float acc = tap(x0,   y0,   (1-lx)*(1-ly)) + tap(x0+1, y0,   lx*(1-ly))
            + tap(x0,   y0+1, (1-lx)*ly)     + tap(x0+1, y0+1, lx*ly);
  qc[row * 256 + c] = acc * g1[row * 256 + c];
}

// sine positional embedding * sigmoid gate; row = q*64 + b*8 + h
__global__ __launch_bounds__(128) void sine_k(
    const float* __restrict__ qpos, const float* __restrict__ gate,
    float* __restrict__ qpe)
{
  const long row = blockIdx.x;
  int t = threadIdx.x;
  int k = t & 63, part = t >> 6;          // part 0: from y, part 1: from x
  float px = qpos[row*2], py = qpos[row*2 + 1];
  float pv = part ? px : py;
  float freq = powf(10000.0f, (float)k * (1.0f / 64.0f));
  float v = pv * 6.283185307179586f / freq;
  int d = part * 128 + 2 * k;
  qpe[row*256 + d]     = sinf(v) * gate[row*256 + d];
  qpe[row*256 + d + 1] = cosf(v) * gate[row*256 + d + 1];
}

// reduce ca_k_w over the broadcast head dim: Ws[i][c] = sum_h W[i][h*256+c]
__global__ __launch_bounds__(256) void wsum_k(
    const float* __restrict__ W, float* __restrict__ Ws)
{
  int i = blockIdx.x, c = threadIdx.x;
  float s = 0.0f;
  #pragma unroll
  for (int h = 0; h < 8; h++) s += W[(long)i * 2048 + h * 256 + c];
  Ws[(long)i * 256 + c] = s;
}

// permute pt2_w1 columns (o*49+p) -> (p*64+o)
__global__ __launch_bounds__(256) void w1p_k(
    const float* __restrict__ W, float* __restrict__ O)
{
  int idx = blockIdx.x;  // o*49+p
  int o = idx / 49, p = idx % 49;
  int j = threadIdx.x;
  O[(long)j * 3136 + p * 64 + o] = W[(long)j * 3136 + idx];
}

// ---------------------------------------------------------------------------
// Host side
// ---------------------------------------------------------------------------
static void gemm(hipStream_t s, int abf, int wbf,
                 const void* A, const void* Wt, const float* bias, void* C,
                 int M, int N, int K, long lda, long ldw, long ldc,
                 long bsA, long bsW, long bsC, int batches, int flags, float alpha)
{
  dim3 grd((N + GBN - 1) / GBN, (M + GBM - 1) / GBM, batches);
  dim3 blk(256, 1, 1);
  if (abf) {
    if (wbf) gemm_k<1,1><<<grd, blk, 0, s>>>(A, Wt, bias, C, M, N, K, lda, ldw, ldc, bsA, bsW, bsC, flags, alpha);
    else     gemm_k<1,0><<<grd, blk, 0, s>>>(A, Wt, bias, C, M, N, K, lda, ldw, ldc, bsA, bsW, bsC, flags, alpha);
  } else {
    if (wbf) gemm_k<0,1><<<grd, blk, 0, s>>>(A, Wt, bias, C, M, N, K, lda, ldw, ldc, bsA, bsW, bsC, flags, alpha);
    else     gemm_k<0,0><<<grd, blk, 0, s>>>(A, Wt, bias, C, M, N, K, lda, ldw, ldc, bsA, bsW, bsC, flags, alpha);
  }
}

enum {
  SA_QC_W=0, SA_QC_B, SA_QP_W, SA_QP_B, SA_KC_W, SA_KC_B, SA_KP_W, SA_KP_B,
  SA_V_W, SA_V_B, CA_QC_W, CA_QC_B, CA_KC_W, CA_KC_B, CA_V_W, CA_V_B,
  CA_OUT_W, CA_OUT_B, SA_OUT_W, SA_OUT_B, CA_VP_W, CA_VP_B,
  SA_IN_W, SA_IN_B, CA_Q_W, CA_Q_B, CA_K_W, CA_K_B,
  CA_QS_W1, CA_QS_B1, CA_QS_W2, CA_QS_B2, CA_KS_W1, CA_KS_B1, CA_KS_W2, CA_KS_B2,
  PT1_W, PT1_B, PT2_W1, PT2_B1, PT2_W2, PT2_B2, PT2_W3, PT2_B3, PT2_W4, PT2_B4,
  ATTN1_W, ATTN1_B, ATTN2_W, ATTN2_B, FF1_W, FF1_B, FF2_W, FF2_B,
  N1_G, N1_B, N2_G, N2_B, N3_G, N3_B
};

extern "C" void kernel_launch(void* const* d_in, const int* in_sizes, int n_in,
                              void* d_out, int out_size, void* d_ws, size_t ws_size,
                              hipStream_t stream)
{
  (void)in_sizes; (void)n_in; (void)out_size; (void)ws_size;

  const float* tgt   = (const float*)d_in[0];
  const float* memry = (const float*)d_in[1];
  const float* pos   = (const float*)d_in[2];
  const float* qrb   = (const float*)d_in[3];
  const float* refb  = (const float*)d_in[4];
  const float* gridp = (const float*)d_in[5];
  const unsigned char* kpm = (const unsigned char*)d_in[6];
  const int* mh = (const int*)d_in[7];
  const int* mw = (const int*)d_in[8];
  auto W = [&](int i) { return (const float*)d_in[9 + i]; };

  size_t cur = 0;
  auto alloc = [&](size_t bytes) -> void* {
    void* p = (char*)d_ws + cur;
    cur += (bytes + 255) & ~(size_t)255;
    return p;
  };

  // ---- persistent buffers ----
  float* vr    = (float*)alloc(32 * 4);
  float* tgt1  = (float*)alloc((size_t)614400 * 4);
  float* tgt2  = (float*)alloc((size_t)614400 * 4);
  float* ptsxy = (float*)alloc((size_t)38400 * 4);
  float* qpos  = (float*)alloc((size_t)38400 * 4);
  float* gp    = (float*)alloc((size_t)38400 * 4);
  float* gs2   = (float*)alloc((size_t)38400 * 4);
  float* qfull = (float*)alloc((size_t)2400 * 2048 * 4);
  float* oca   = (float*)alloc((size_t)614400 * 4);
  const size_t mark0 = cur;

  vr_k<<<1, 32, 0, stream>>>(kpm, mh, mw, vr);

  // ================= Stage 1: self-attention =================
  {
    float* qs  = (float*)alloc((size_t)614400 * 4);
    float* ks_ = (float*)alloc((size_t)614400 * 4);
    float* vs  = (float*)alloc((size_t)614400 * 4);
    float* qh  = (float*)alloc((size_t)614400 * 4);
    float* kh  = (float*)alloc((size_t)614400 * 4);
    float* vh  = (float*)alloc((size_t)614400 * 4);
    float* ssa = (float*)alloc((size_t)64 * 300 * 320 * 4);
    bf16*  asa = (bf16*) alloc((size_t)64 * 300 * 320 * 2);
    bf16*  vts = (bf16*) alloc((size_t)64 * 32 * 320 * 2);
    float* osa = (float*)alloc((size_t)614400 * 4);
    float* opj = (float*)alloc((size_t)614400 * 4);

    gemm(stream,0,0, tgt, W(SA_QC_W), W(SA_QC_B), qs,  2400,256,256, 256,256,256, 0,0,0, 1, 0, 1.0f);
    gemm(stream,0,0, qrb, W(SA_QP_W), W(SA_QP_B), qs,  2400,256,256, 256,256,256, 0,0,0, 1, 4, 1.0f);
    gemm(stream,0,0, tgt, W(SA_KC_W), W(SA_KC_B), ks_, 2400,256,256, 256,256,256, 0,0,0, 1, 0, 1.0f);
    gemm(stream,0,0, qrb, W(SA_KP_W), W(SA_KP_B), ks_, 2400,256,256, 256,256,256, 0,0,0, 1, 4, 1.0f);
    gemm(stream,0,0, tgt, W(SA_V_W),  W(SA_V_B),  vs,  2400,256,256, 256,256,256, 0,0,0, 1, 0, 1.0f);
    const float* inw = W(SA_IN_W); const float* inb = W(SA_IN_B);
    gemm(stream,0,0, qs,  inw,           inb,       qh, 2400,256,256, 256,256,256, 0,0,0, 1, 0, 0.17677669529663687f);
    gemm(stream,0,0, ks_, inw + 256*256, inb + 256, kh, 2400,256,256, 256,256,256, 0,0,0, 1, 0, 1.0f);
    gemm(stream,0,0, vs,  inw + 512*256, inb + 512, vh, 2400,256,256, 256,256,256, 0,0,0, 1, 0, 1.0f);
    // scores[b,h]: 300x300, K=32 (head dim); batch offset collapses to 32*bi
    gemm(stream,0,0, qh, kh, nullptr, ssa, 300,300,32, 2048,2048,320, 32,32,(long)300*320, 64, 0, 1.0f);
    softmax_sa_k<<<19200, 256, 0, stream>>>(ssa, asa);
    vtr_k<<<64, 256, 0, stream>>>(vh, vts, 300, 320);
    gemm(stream,1,1, asa, vts, nullptr, osa, 300,32,320, 320,320,2048, (long)300*320,(long)32*320,32, 64, 0, 1.0f);
    gemm(stream,0,0, osa, W(SA_OUT_W), W(SA_OUT_B), opj, 2400,256,256, 256,256,256, 0,0,0, 1, 0, 1.0f);
    ln_k<<<2400, 256, 0, stream>>>(tgt, opj, W(N1_G), W(N1_B), tgt1);
  }
  cur = mark0;

  // ================= Stage 2: ROI + point MLP + gated sampling =================
  {
    bf16*  roib = (bf16*) alloc((size_t)117600 * 256 * 2);
    float* pts1 = (float*)alloc((size_t)117600 * 64 * 4);
    float* w1p  = (float*)alloc((size_t)256 * 3136 * 4);
    float* pts2 = (float*)alloc((size_t)2400 * 256 * 4);
    float* pts3 = (float*)alloc((size_t)2400 * 512 * 4);
    float* pts4 = (float*)alloc((size_t)2400 * 512 * 4);
    float* pts5 = (float*)alloc((size_t)2400 * 32 * 4);
    float* g1   = (float*)alloc((size_t)19200 * 256 * 4);
    float* qcb  = (float*)alloc((size_t)19200 * 256 * 4);

    roi_k<<<117600, 256, 0, stream>>>(memry, refb, vr, mh, mw, roib);
    gemm(stream,1,0, roib, W(PT1_W), W(PT1_B), pts1, 117600,64,256, 256,256,64, 0,0,0, 1, 1, 1.0f);
    w1p_k<<<3136, 256, 0, stream>>>(W(PT2_W1), w1p);
    gemm(stream,0,0, pts1, w1p, W(PT2_B1), pts2, 2400,256,3136, 3136,3136,256, 0,0,0, 1, 1, 1.0f);
    gemm(stream,0,0, pts2, W(PT2_W2), W(PT2_B2), pts3, 2400,512,256, 256,256,512, 0,0,0, 1, 1, 1.0f);
    gemm(stream,0,0, pts3, W(PT2_W3), W(PT2_B3), pts4, 2400,512,512, 512,512,512, 0,0,0, 1, 1, 1.0f);
    gemm(stream,0,0, pts4, W(PT2_W4), W(PT2_B4), pts5, 2400,32,512, 512,512,32, 0,0,0, 1, 0, 1.0f);
    points_k<<<75, 256, 0, stream>>>(pts5, refb, vr, mh, mw, ptsxy, qpos, gp, gs2);
    gemm(stream,0,0, tgt1, W(ATTN1_W), W(ATTN1_B), g1, 2400,2048,256, 256,256,2048, 0,0,0, 1, 2, 1.0f);
    sample_k<<<19200, 256, 0, stream>>>(roib, ptsxy, g1, qcb);
    gemm(stream,0,0, qcb, W(CA_QC_W), W(CA_QC_B), qfull, 19200,256,256, 256,256,256, 0,0,0, 1, 0, 1.0f);
  }
  cur = mark0;

  // ================= Stage 3a: gated sine embedding -> qse, qfull += qse =================
  {
    float* g2   = (float*)alloc((size_t)19200 * 256 * 4);
    float* qpe  = (float*)alloc((size_t)19200 * 256 * 4);
    float* qseh = (float*)alloc((size_t)19200 * 256 * 4);
    gemm(stream,0,0, tgt1, W(ATTN2_W), W(ATTN2_B), g2, 2400,2048,256, 256,256,2048, 0,0,0, 1, 2, 1.0f);
    sine_k<<<19200, 128, 0, stream>>>(qpos, g2, qpe);
    gemm(stream,0,0, qpe,  W(CA_QS_W1), W(CA_QS_B1), qseh,  19200,256,256, 256,256,256, 0,0,0, 1, 1, 1.0f);
    gemm(stream,0,0, qseh, W(CA_QS_W2), W(CA_QS_B2), qfull, 19200,256,256, 256,256,256, 0,0,0, 1, 4, 1.0f);
  }
  cur = mark0;

  // ================= Stage 3b: cross-attention + FFN =================
  {
    float* kc   = (float*)alloc((size_t)8192 * 256 * 4);
    float* kph  = (float*)alloc((size_t)8192 * 256 * 4);
    float* wsum = (float*)alloc((size_t)2048 * 256 * 4);
    float* vmv  = (float*)alloc((size_t)8192 * 256 * 4);
    float* vh2  = (float*)alloc((size_t)8192 * 256 * 4);
    bf16*  vtc  = (bf16*) alloc((size_t)64 * 32 * 1024 * 2);
    bf16*  qhc  = (bf16*) alloc((size_t)2400 * 2048 * 2);
    float* sca  = (float*)alloc((size_t)64 * 300 * 1024 * 4);
    bf16*  aca  = (bf16*) alloc((size_t)64 * 300 * 1024 * 2);
    const size_t markK = cur;
    bf16*  khc  = (bf16*) alloc((size_t)8192 * 2048 * 2);

    gemm(stream,0,0, memry, W(CA_KC_W),  W(CA_KC_B),  kc,  8192,256,256, 256,256,256, 0,0,0, 1, 0, 1.0f);
    gemm(stream,0,0, pos,   W(CA_KS_W1), W(CA_KS_B1), kph, 8192,256,256, 256,256,256, 0,0,0, 1, 1, 1.0f);
    gemm(stream,0,0, kph,   W(CA_KS_W2), W(CA_KS_B2), kc,  8192,256,256, 256,256,256, 0,0,0, 1, 4, 1.0f);
    wsum_k<<<2048, 256, 0, stream>>>(W(CA_K_W), wsum);
    gemm(stream,0,0, memry, W(CA_V_W),  W(CA_V_B),  vmv, 8192,256,256, 256,256,256, 0,0,0, 1, 0, 1.0f);
    gemm(stream,0,0, vmv,   W(CA_VP_W), W(CA_VP_B), vh2, 8192,256,256, 256,256,256, 0,0,0, 1, 0, 1.0f);
    vtr_k<<<64, 256, 0, stream>>>(vh2, vtc, 1024, 1024);
    // q projection (alpha = 256^-0.5 = 1/16), bf16 output
    gemm(stream,0,0, qfull, W(CA_Q_W), W(CA_Q_B), qhc, 2400,2048,2048, 2048,2048,2048, 0,0,0, 1, 8, 0.0625f);
    // k projection via head-broadcast reduction (8192x2048x256 instead of x2048)
    gemm(stream,0,0, kc, wsum, W(CA_K_B), khc, 8192,2048,256, 256,256,2048, 0,0,0, 1, 8, 1.0f);
    // logits[b,h]: 300x1024, K=256
    gemm(stream,1,1, qhc, khc, nullptr, sca, 300,1024,256, 16384,16384,1024, 256,256,(long)300*1024, 64, 0, 1.0f);
    cur = markK;  // khc dead (stream-ordered reuse is safe)
    softmax_ca_k<<<19200, 256, 0, stream>>>(sca, gp, gs2, gridp, kpm, aca);
    gemm(stream,1,1, aca, vtc, nullptr, oca, 300,32,1024, 1024,1024,2048, (long)300*1024,(long)32*1024,32, 64, 0, 1.0f);
    float* opj2 = (float*)alloc((size_t)614400 * 4);
    gemm(stream,0,0, oca, W(CA_OUT_W), W(CA_OUT_B), opj2, 2400,256,256, 256,256,256, 0,0,0, 1, 0, 1.0f);
    ln_k<<<2400, 256, 0, stream>>>(tgt1, opj2, W(N2_G), W(N2_B), tgt2);
    cur = mark0;
    float* ffh = (float*)alloc((size_t)2400 * 2048 * 4);
    float* ffo = (float*)alloc((size_t)614400 * 4);
    gemm(stream,0,0, tgt2, W(FF1_W), W(FF1_B), ffh, 2400,2048,256, 256,256,2048, 0,0,0, 1, 1, 1.0f);
    gemm(stream,0,0, ffh,  W(FF2_W), W(FF2_B), ffo, 2400,256,2048, 2048,2048,256, 0,0,0, 1, 0, 1.0f);
    ln_k<<<2400, 256, 0, stream>>>(tgt2, ffo, W(N3_G), W(N3_B), (float*)d_out);
  }
}
